// LocalSelfAttention_2817498546292
// MI455X (gfx1250) — matrix operand order repaired and checked by reading, not compiled, not run on hardware
//
#include <hip/hip_runtime.h>

typedef __attribute__((ext_vector_type(16))) _Float16 v16h;
typedef __attribute__((ext_vector_type(8)))  _Float16 v8h;
typedef __attribute__((ext_vector_type(8)))  float    v8f;

#define HIDW   768
#define NSEQ   2048
#define NHEAD  12
#define HDIM   64
#define WHALF  128
#define PW     272     /* padded rel-bucket width (17*16); real P = 257 */
#define BAND   288     /* LDS score columns (18*16), covers band width 272 */
#define SSTR   292     /* LDS row stride (floats), conflict padding */
#define NEG_LV (-3.4028234663852886e38f)

static __device__ inline v8f wmma16(v16h a, v16h b, v8f c) {
  return __builtin_amdgcn_wmma_f32_16x16x32_f16(false, a, false, b, (short)0, c,
                                                false, false);
}

// A fragment (16x32, f16): lane L<16 holds row L, K = k0+[0..15];
// lane L+16 holds row L, K = k0+16+[0..15].
static __device__ inline v16h load_a_frag(const _Float16* __restrict__ A, int lda,
                                          int mbase, int k0) {
  int lane = threadIdx.x & 31;
  const _Float16* row =
      A + (size_t)(mbase + (lane & 15)) * lda + k0 + (lane >> 4) * 16;
  v8h lo = *(const v8h*)row;
  v8h hi = *(const v8h*)(row + 8);
  v16h a;
#pragma unroll
  for (int e = 0; e < 8; ++e) { a[e] = lo[e]; a[8 + e] = hi[e]; }
  return a;
}

// B fragment (32x16) from row-major B^T (i.e. [n, k] storage): lane L holds
// column n = nbase + (L&15), K = k0 + (L>>4)*16 + [0..15]  (contiguous load).
static __device__ inline v16h load_bt_frag(const _Float16* __restrict__ BT, int ldb,
                                           int nbase, int k0, int nmax) {
  int lane = threadIdx.x & 31;
  int n = nbase + (lane & 15);
  n = min(max(n, 0), nmax);
  const _Float16* row = BT + (size_t)n * ldb + k0 + (lane >> 4) * 16;
  return *(const v16h*)row;
}

// Async global->LDS 16-byte copy (CDNA5, ASYNCcnt-tracked).
static __device__ inline void async_copy_b128(unsigned lds_off, const void* gptr) {
  asm volatile("global_load_async_to_lds_b128 %0, %1, off"
               :: "v"(lds_off), "v"((unsigned long long)(uintptr_t)gptr)
               : "memory");
}
static __device__ inline void wait_async0() {
  asm volatile("s_wait_asynccnt 0" ::: "memory");
}

// ---------------------------------------------------------------- utilities
__global__ void cvt_kernel(const float* __restrict__ in, _Float16* __restrict__ out,
                           int n) {
  int i = blockIdx.x * blockDim.x + threadIdx.x;
  if (i < n) out[i] = (_Float16)in[i];
}

// position_{key,query}[h,d,p] (f32, P=257) -> [h, p(272, zero-pad), d] f16
__global__ void posT_kernel(const float* __restrict__ pk, const float* __restrict__ pq,
                            _Float16* __restrict__ pkT, _Float16* __restrict__ pqT) {
  int e = blockIdx.x * blockDim.x + threadIdx.x;
  if (e >= NHEAD * PW * HDIM) return;
  int d = e & 63;
  int p = (e >> 6) % PW;
  int h = e / (PW * HDIM);
  float vk = 0.f, vq = 0.f;
  if (p < 2 * WHALF + 1) {
    size_t src = ((size_t)h * HDIM + d) * (2 * WHALF + 1) + p;
    vk = pk[src];
    vq = pq[src];
  }
  pkT[e] = (_Float16)vk;
  pqT[e] = (_Float16)vq;
}

// --------------------------------------------------------------- QKV GEMMs
// q/k/v[o] = sum_c X[g,c] * W[o,c]; stored as f16 [B,H,N,64]
__global__ void __launch_bounds__(256)
qkv_kernel(const _Float16* __restrict__ X, const _Float16* __restrict__ Wq,
           const _Float16* __restrict__ Wk, const _Float16* __restrict__ Wv,
           _Float16* __restrict__ qo, _Float16* __restrict__ ko,
           _Float16* __restrict__ vo) {
  int wave = threadIdx.x >> 5, lane = threadIdx.x & 31;
  int n0 = (blockIdx.x * 8 + wave) * 16;   // output column tile (0..767)
  int m0 = blockIdx.y * 16;                // row tile (0..4095)
  const _Float16* W = (blockIdx.z == 0) ? Wq : (blockIdx.z == 1) ? Wk : Wv;
  _Float16* dst = (blockIdx.z == 0) ? qo : (blockIdx.z == 1) ? ko : vo;
  v8f acc = {};
  for (int k0 = 0; k0 < HIDW; k0 += 32) {
    v16h a = load_a_frag(X, HIDW, m0, k0);
    v16h b = load_bt_frag(W, HIDW, n0, k0, HIDW - 1);
    acc = wmma16(a, b, acc);
  }
  int o = n0 + (lane & 15);
  int h = o >> 6, d = o & 63;
#pragma unroll
  for (int r = 0; r < 8; ++r) {
    int g = m0 + (lane >> 4) * 8 + r;      // global row (b*N + i)
    int b = g >> 11, i = g & (NSEQ - 1);
    dst[(((size_t)b * NHEAD + h) * NSEQ + i) * HDIM + d] = (_Float16)acc[r];
  }
}

// ------------------------------------------------- c2p / p2c rel-pos GEMMs
// z=0: c2p[g,p] = sum_d q[g,d]*posKeyT[h,p,d];  z=1: p2c from k/posQryT
__global__ void __launch_bounds__(128)
relpos_kernel(const _Float16* __restrict__ q, const _Float16* __restrict__ k,
              const _Float16* __restrict__ pkT, const _Float16* __restrict__ pqT,
              _Float16* __restrict__ c2p, _Float16* __restrict__ p2c) {
  int wave = threadIdx.x >> 5, lane = threadIdx.x & 31;
  int pt = blockIdx.x * 4 + wave;          // p-tile 0..16
  if (pt >= PW / 16) return;               // whole-wave uniform exit
  int rowblk = blockIdx.y;                 // 0..(B*H*N/16 - 1)
  int z = blockIdx.z;
  const _Float16* A = z ? k : q;
  const _Float16* PT = z ? pqT : pkT;
  _Float16* O = z ? p2c : c2p;
  int h = (rowblk >> 7) % NHEAD;           // 128 row-blocks per (b,h)
  int m0 = rowblk * 16;
  const _Float16* PTh = PT + (size_t)h * PW * HDIM;
  v16h a0 = load_a_frag(A, HDIM, m0, 0);
  v16h a1 = load_a_frag(A, HDIM, m0, 32);
  v16h b0 = load_bt_frag(PTh, HDIM, pt * 16, 0, PW - 1);
  v16h b1 = load_bt_frag(PTh, HDIM, pt * 16, 32, PW - 1);
  v8f acc = {};
  acc = wmma16(a0, b0, acc);
  acc = wmma16(a1, b1, acc);
  int n = lane & 15;
#pragma unroll
  for (int r = 0; r < 8; ++r) {
    int g = m0 + (lane >> 4) * 8 + r;
    O[(size_t)g * PW + pt * 16 + n] = (_Float16)acc[r];
  }
}

// -------------------------------------------------------- banded attention
__global__ void __launch_bounds__(128)
attn_kernel(const _Float16* __restrict__ q, const _Float16* __restrict__ k,
            const _Float16* __restrict__ v, const _Float16* __restrict__ c2p,
            const _Float16* __restrict__ p2c, const unsigned char* __restrict__ am,
            float* __restrict__ proba, _Float16* __restrict__ ctx) {
  __shared__ float sc[16][SSTR];
  __shared__ float rowmax[16], rowsum[16], red[16][8];
  __shared__ _Float16 vlds[BAND][HDIM];    // V tile staged via async copies

  int ib = blockIdx.x;                 // 16-row query block
  int bh = blockIdx.y;                 // b*H + h
  int b = bh / NHEAD, h = bh % NHEAD;
  int i0 = ib * 16;
  int jstart = i0 - WHALF;
  int tid = threadIdx.x, wave = tid >> 5, lane = tid & 31;

  const _Float16* qbh = q + (size_t)bh * NSEQ * HDIM;
  const _Float16* kbh = k + (size_t)bh * NSEQ * HDIM;
  const _Float16* vbh = v + (size_t)bh * NSEQ * HDIM;
  const _Float16* c2pbh = c2p + (size_t)bh * NSEQ * PW;
  const _Float16* p2cbh = p2c + (size_t)bh * NSEQ * PW;
  const unsigned char* amb = am + (size_t)b * NSEQ;

  // ---- pass 0: kick off async V-tile prefetch (288 rows x 128B = 2304 x 16B)
  {
    unsigned vbase = (unsigned)(uintptr_t)&vlds[0][0];
#pragma unroll
    for (int t = 0; t < (BAND * HDIM * 2 / 16) / 128; ++t) {   // 18 chunks/thread
      int cidx = tid + t * 128;
      int row = cidx >> 3;                 // 8 x 16B chunks per 128B row
      int c16 = cidx & 7;
      int j = min(max(jstart + row, 0), NSEQ - 1);   // clamp: prob=0 there
      async_copy_b128(vbase + row * 128 + c16 * 16,
                      vbh + (size_t)j * HDIM + c16 * 8);
    }
  }

  // ---- pass 1: scores for 18 j-tiles, striped over 4 waves
  v16h a0 = load_a_frag(qbh, HDIM, i0, 0);
  v16h a1 = load_a_frag(qbh, HDIM, i0, 32);
  int nl = lane & 15, rbase = (lane >> 4) * 8;
  for (int t = wave; t < BAND / 16; t += 4) {
    int jt = jstart + t * 16;
    v16h b0 = load_bt_frag(kbh, HDIM, jt, 0, NSEQ - 1);
    v16h b1 = load_bt_frag(kbh, HDIM, jt, 32, NSEQ - 1);
    v8f acc = {};
    acc = wmma16(a0, b0, acc);
    acc = wmma16(a1, b1, acc);
    int j = jt + nl;
#pragma unroll
    for (int r = 0; r < 8; ++r) {
      int i = i0 + rbase + r;
      float s;
      if (j >= 0 && j < NSEQ) {
        int pr = j - i + WHALF;
        int pc = min(max(pr, 0), 2 * WHALF);                 // c2p bucket
        int p2 = min(max(2 * WHALF - pr, 0), 2 * WHALF);     // p2c bucket
        s = acc[r] + (float)c2pbh[(size_t)i * PW + pc] +
            (float)p2cbh[(size_t)j * PW + p2];
        bool band = (j - i <= WHALF) && (i - j <= WHALF);
        if (!(band && amb[i] && amb[j])) s += NEG_LV;
      } else {
        s = NEG_LV;                                          // j outside [0,N)
      }
      sc[rbase + r][j - jstart] = s;
    }
  }
  __syncthreads();

  // ---- pass 2: row softmax (8 threads per row)
  int row = tid >> 3, sub = tid & 7;
  float mx = -INFINITY;
  for (int c = sub; c < BAND; c += 8) mx = fmaxf(mx, sc[row][c]);
  red[row][sub] = mx;
  __syncthreads();
  if (sub == 0) {
    float m = red[row][0];
    for (int s = 1; s < 8; ++s) m = fmaxf(m, red[row][s]);
    rowmax[row] = m;
  }
  __syncthreads();
  float rm = rowmax[row];
  float sum = 0.f;
  for (int c = sub; c < BAND; c += 8) {
    float e = __expf(sc[row][c] - rm);
    sc[row][c] = e;
    sum += e;
  }
  red[row][sub] = sum;
  __syncthreads();
  if (sub == 0) {
    float s = 0.f;
    for (int t = 0; t < 8; ++t) s += red[row][t];
    rowsum[row] = s;
  }
  __syncthreads();
  float inv = 1.0f / rowsum[row];
  for (int c = sub; c < BAND; c += 8) sc[row][c] *= inv;
  __syncthreads();

  // ---- pass 2b: write 16 FULL rows of proba (zeros outside band), coalesced.
  // This block owns rows [i0, i0+16) of (b,h), so no separate zero-fill pass.
  float* prow = proba + ((size_t)bh * NSEQ + i0) * NSEQ;
  for (int e = tid; e < 16 * NSEQ; e += 128) {
    int m = e >> 11;                       // NSEQ == 2048
    int j = e & (NSEQ - 1);
    int c = j - jstart;
    float val = (c >= 0 && c < BAND) ? sc[m][c] : 0.f;
    prow[(size_t)m * NSEQ + j] = val;
  }

  // ---- pass 3: context = P(16x288) @ V(288x64); wave -> 16-wide d tile
  wait_async0();                           // this wave's V chunks are in LDS
  __syncthreads();                         // ... and everyone else's
  int d0 = wave * 16;
  v8f acc = {};
  int mrow = lane & 15, kh = (lane >> 4) * 16;
  int dcol = d0 + (lane & 15), kb = (lane >> 4) * 16;
  for (int ks = 0; ks < BAND; ks += 32) {
    v16h pa;
#pragma unroll
    for (int e = 0; e < 8; ++e) {
      pa[e] = (_Float16)sc[mrow][ks + kh + e];
      pa[8 + e] = (_Float16)sc[mrow][ks + kh + 8 + e];
    }
    v16h vb;
#pragma unroll
    for (int e = 0; e < 16; ++e) vb[e] = vlds[ks + kb + e][dcol];
    acc = wmma16(pa, vb, acc);
  }
#pragma unroll
  for (int r = 0; r < 8; ++r) {
    int i = i0 + (lane >> 4) * 8 + r;
    ctx[((size_t)b * NSEQ + i) * HIDW + h * HDIM + d0 + (lane & 15)] =
        (_Float16)acc[r];
  }
}

// ------------------------------------------------- output projection + bias
__global__ void __launch_bounds__(256)
oproj_kernel(const _Float16* __restrict__ ctx, const _Float16* __restrict__ Wo,
             const float* __restrict__ bo, float* __restrict__ out) {
  int wave = threadIdx.x >> 5, lane = threadIdx.x & 31;
  int n0 = (blockIdx.x * 8 + wave) * 16;
  int m0 = blockIdx.y * 16;
  v8f acc = {};
  for (int k0 = 0; k0 < HIDW; k0 += 32) {
    v16h a = load_a_frag(ctx, HIDW, m0, k0);
    v16h b = load_bt_frag(Wo, HIDW, n0, k0, HIDW - 1);
    acc = wmma16(a, b, acc);
  }
  int n = n0 + (lane & 15);
  float bias = bo[n];
#pragma unroll
  for (int r = 0; r < 8; ++r) {
    int g = m0 + (lane >> 4) * 8 + r;
    out[(size_t)g * HIDW + n] = acc[r] + bias;
  }
}

extern "C" void kernel_launch(void* const* d_in, const int* in_sizes, int n_in,
                              void* d_out, int out_size, void* d_ws, size_t ws_size,
                              hipStream_t stream) {
  const float* hs = (const float*)d_in[0];
  const unsigned char* am = (const unsigned char*)d_in[1];  // jnp.bool_
  const float* Wq = (const float*)d_in[2];
  const float* Wk = (const float*)d_in[3];
  const float* Wv = (const float*)d_in[4];
  const float* pq = (const float*)d_in[5];  // position_query
  const float* pk = (const float*)d_in[6];  // position_key
  const float* Wo = (const float*)d_in[7];
  const float* bo = (const float*)d_in[8];

  const size_t nBN = (size_t)2 * NSEQ;                // 4096 rows
  const size_t nX = nBN * HIDW;                       // 3,145,728
  const size_t nW = (size_t)HIDW * HIDW;              // 589,824
  const size_t nQKV = (size_t)2 * NHEAD * NSEQ * HDIM;
  const size_t nPT = (size_t)NHEAD * PW * HDIM;       // 208,896
  const size_t nREL = (size_t)2 * NHEAD * NSEQ * PW;  // 13,369,344

  char* w = (char*)d_ws;
  auto alloc = [&](size_t bytes) {
    char* p = w;
    w += (bytes + 255) & ~(size_t)255;
    return p;
  };
  _Float16* Xh = (_Float16*)alloc(nX * 2);
  _Float16* Wqh = (_Float16*)alloc(nW * 2);
  _Float16* Wkh = (_Float16*)alloc(nW * 2);
  _Float16* Wvh = (_Float16*)alloc(nW * 2);
  _Float16* Woh = (_Float16*)alloc(nW * 2);
  _Float16* qws = (_Float16*)alloc(nQKV * 2);
  _Float16* kws = (_Float16*)alloc(nQKV * 2);
  _Float16* vws = (_Float16*)alloc(nQKV * 2);
  _Float16* pkT = (_Float16*)alloc(nPT * 2);
  _Float16* pqT = (_Float16*)alloc(nPT * 2);
  _Float16* c2pw = (_Float16*)alloc(nREL * 2);
  _Float16* p2cw = (_Float16*)alloc(nREL * 2);
  _Float16* ctxh = (_Float16*)alloc(nX * 2);

  float* out0 = (float*)d_out;
  float* proba = out0 + nX;                           // [B,H,N,N]

  cvt_kernel<<<(int)(nX + 255) / 256, 256, 0, stream>>>(hs, Xh, (int)nX);
  cvt_kernel<<<(int)(nW + 255) / 256, 256, 0, stream>>>(Wq, Wqh, (int)nW);
  cvt_kernel<<<(int)(nW + 255) / 256, 256, 0, stream>>>(Wk, Wkh, (int)nW);
  cvt_kernel<<<(int)(nW + 255) / 256, 256, 0, stream>>>(Wv, Wvh, (int)nW);
  cvt_kernel<<<(int)(nW + 255) / 256, 256, 0, stream>>>(Wo, Woh, (int)nW);
  posT_kernel<<<(int)(nPT + 255) / 256, 256, 0, stream>>>(pk, pq, pkT, pqT);

  qkv_kernel<<<dim3(6, 256, 3), 256, 0, stream>>>(Xh, Wqh, Wkh, Wvh, qws, kws, vws);
  relpos_kernel<<<dim3(5, 3072, 2), 128, 0, stream>>>(qws, kws, pkT, pqT, c2pw, p2cw);
  attn_kernel<<<dim3(128, 24), 128, 0, stream>>>(qws, kws, vws, c2pw, p2cw, am,
                                                 proba, ctxh);
  oproj_kernel<<<dim3(6, 256), 256, 0, stream>>>(ctxh, Woh, bo, out0);
}
